// MultiHeadAttention_78451872628881
// MI455X (gfx1250) — matrix operand order repaired and checked by reading, not compiled, not run on hardware
//
#include <hip/hip_runtime.h>
#include <hip/hip_bf16.h>
#include <stdint.h>

// ---------------------------------------------------------------------------
// MHA forward for MI455X (gfx1250), wave32 + v_wmma_f32_16x16x32_f16.
//   B=2, S=2048, D=1024, H=16, dk=64
// ---------------------------------------------------------------------------

typedef _Float16 h16;
typedef __attribute__((ext_vector_type(16))) _Float16 v16h;
typedef __attribute__((ext_vector_type(8)))  _Float16 v8h;
typedef __attribute__((ext_vector_type(8)))  float    v8f;

#define WMMA16(a, b, c) \
  __builtin_amdgcn_wmma_f32_16x16x32_f16(false, (a), false, (b), (short)0, (c), false, false)

#define BATCH  2
#define SEQ    2048
#define DMODEL 1024
#define NH     16
#define DK     64
#define MROWS  4096  // BATCH*SEQ

// ---- fragment loaders (layouts per CDNA5 ISA 7.12.2) ----------------------
// A 16x32 f16: lane l holds row (l&15); K = (l>>4)*8 + {0..7} and +16+{0..7}
__device__ __forceinline__ v16h load_a(const h16* s, int lane, int stride) {
  const int r  = lane & 15;
  const int kb = (lane >> 4) * 8;
  v8h lo = *(const v8h*)(s + (size_t)r * stride + kb);
  v8h hi = *(const v8h*)(s + (size_t)r * stride + kb + 16);
  return __builtin_shufflevector(lo, hi, 0, 1, 2, 3, 4, 5, 6, 7,
                                 8, 9, 10, 11, 12, 13, 14, 15);
}
// B 32x16 f16 from an [N][K] row-major tile: lane l holds col (l&15);
// K = (l>>4)*8 + {0..7} and +16+{0..7} (same order as load_a).
__device__ __forceinline__ v16h load_b(const h16* s, int lane, int stride) {
  const int n  = lane & 15;
  const int kb = (lane >> 4) * 8;
  v8h lo = *(const v8h*)(s + (size_t)n * stride + kb);
  v8h hi = *(const v8h*)(s + (size_t)n * stride + kb + 16);
  return __builtin_shufflevector(lo, hi, 0, 1, 2, 3, 4, 5, 6, 7,
                                 8, 9, 10, 11, 12, 13, 14, 15);
}

// ---------------------------------------------------------------------------
// Kernel 1: QKV projection.  Y = x @ W^T  (M=4096, N=1024, K=1024)
// grid = (32, 16, 3); block = 128 (4 waves).
// Block tile 128(M) x 64(N), K step 64; wave tile 32x64 => 16 WMMA / barrier.
// mode 0: Q (scaled 1/8) -> [B,H,S,dk]; 1: K -> [B,H,S,dk]; 2: V^T -> [B,H,dk,S]
// ---------------------------------------------------------------------------
__global__ __launch_bounds__(128) void proj_qkv(
    const float* __restrict__ x,
    const float* __restrict__ Wq, const float* __restrict__ Wk,
    const float* __restrict__ Wv,
    h16* __restrict__ qo, h16* __restrict__ ko, h16* __restrict__ vto) {
  __shared__ h16 sA[128 * 64];  // 16 KB
  __shared__ h16 sB[64 * 64];   //  8 KB
  const int mt = blockIdx.x, nt = blockIdx.y, mode = blockIdx.z;
  const float* W = (mode == 0) ? Wq : ((mode == 1) ? Wk : Wv);
  const int tid = threadIdx.x;
  const int lane = tid & 31, wave = tid >> 5;

  v8f accA[4] = {v8f{}, v8f{}, v8f{}, v8f{}};
  v8f accB[4] = {v8f{}, v8f{}, v8f{}, v8f{}};

  for (int k0 = 0; k0 < DMODEL; k0 += 64) {
    __syncthreads();
    // A tile: x[mt*128 .. +127][k0 .. +63] fp32 -> fp16 (2048 float4, 16/thread)
#pragma unroll
    for (int j = 0; j < 16; ++j) {
      int idx = tid + j * 128;
      int r = idx >> 4, c4 = idx & 15;
      float4 f = *(const float4*)(x + (size_t)(mt * 128 + r) * DMODEL + k0 + c4 * 4);
      h16* d = sA + r * 64 + c4 * 4;
      d[0] = (h16)f.x; d[1] = (h16)f.y; d[2] = (h16)f.z; d[3] = (h16)f.w;
    }
    // B tile: W[nt*64 .. +63][k0 .. +63] (rows of W are N, cols K)
#pragma unroll
    for (int j = 0; j < 8; ++j) {
      int idx = tid + j * 128;
      int r = idx >> 4, c4 = idx & 15;
      float4 f = *(const float4*)(W + (size_t)(nt * 64 + r) * DMODEL + k0 + c4 * 4);
      h16* d = sB + r * 64 + c4 * 4;
      d[0] = (h16)f.x; d[1] = (h16)f.y; d[2] = (h16)f.z; d[3] = (h16)f.w;
    }
    __syncthreads();

    const h16* aBase = sA + (size_t)(wave * 32) * 64;
    v16h a00 = load_a(aBase, lane, 64);            // rows 0-15,  k 0-31
    v16h a01 = load_a(aBase + 32, lane, 64);       // rows 0-15,  k 32-63
    v16h a10 = load_a(aBase + 16 * 64, lane, 64);  // rows 16-31, k 0-31
    v16h a11 = load_a(aBase + 16 * 64 + 32, lane, 64);
#pragma unroll
    for (int n = 0; n < 4; ++n) {
      v16h b0 = load_b(sB + n * 16 * 64, lane, 64);
      v16h b1 = load_b(sB + n * 16 * 64 + 32, lane, 64);
      accA[n] = WMMA16(a00, b0, accA[n]);
      accA[n] = WMMA16(a01, b1, accA[n]);
      accB[n] = WMMA16(a10, b0, accB[n]);
      accB[n] = WMMA16(a11, b1, accB[n]);
    }
  }

  const int hi8 = (lane & 16) ? 8 : 0;
  const int col = lane & 15;
#pragma unroll
  for (int g = 0; g < 2; ++g) {
    const v8f* acc = g ? accB : accA;
#pragma unroll
    for (int v = 0; v < 8; ++v) {
      int rowg = mt * 128 + wave * 32 + g * 16 + hi8 + v;
      int b = rowg >> 11, s = rowg & (SEQ - 1);
#pragma unroll
      for (int n = 0; n < 4; ++n) {
        int colg = nt * 64 + n * 16 + col;
        int h = colg >> 6, f = colg & (DK - 1);
        float val = acc[n][v];
        if (mode == 0) val *= 0.125f;  // 1/sqrt(dk)
        if (mode == 2)
          vto[(((size_t)b * NH + h) * DK + f) * SEQ + s] = (h16)val;
        else if (mode == 0)
          qo[(((size_t)b * NH + h) * SEQ + s) * DK + f] = (h16)val;
        else
          ko[(((size_t)b * NH + h) * SEQ + s) * DK + f] = (h16)val;
      }
    }
  }
}

// ---------------------------------------------------------------------------
// Kernel 2: fused causal attention (flash-style, online softmax).
// grid = B*H*(S/64) blocks, 128 threads (4 waves); wave owns 16 query rows.
// Key tile = 64 -> 8 WMMA (QK^T) + 8 WMMA (P.V) per barrier per wave.
// ---------------------------------------------------------------------------
__global__ __launch_bounds__(128) void attn(
    const h16* __restrict__ q, const h16* __restrict__ k,
    const h16* __restrict__ vt, h16* __restrict__ outp) {
  __shared__ h16 sK[64 * 64];      // [key][feat]   8 KB
  __shared__ h16 sV[64 * 64];      // [feat][key]   8 KB (V already transposed)
  __shared__ h16 sP[4 * 16 * 64];  // per-wave probability tile, 8 KB

  const int tid = threadIdx.x;
  const int lane = tid & 31, wave = tid >> 5;
  const int bid = blockIdx.x;
  const int qt = bid & 31;  // S/64 = 32 query tiles
  const int h  = (bid >> 5) & (NH - 1);
  const int b  = bid >> 9;

  const size_t hb = ((size_t)b * NH + h) * SEQ * DK;
  const h16* qh = q + hb;
  const h16* kh = k + hb;
  const h16* vh = vt + hb;  // [DK][SEQ]

  const int qbase = qt * 64 + wave * 16;
  const int hi8 = (lane & 16) ? 8 : 0;
  const int col = lane & 15;

  // Q fragments: two K-steps of 32 over dk=64
  v16h qa0 = load_a(qh + (size_t)qbase * DK, lane, DK);
  v16h qa1 = load_a(qh + (size_t)qbase * DK + 32, lane, DK);

  v8f o[4] = {v8f{}, v8f{}, v8f{}, v8f{}};
  float m[8], l[8];
#pragma unroll
  for (int v = 0; v < 8; ++v) { m[v] = -3.0e38f; l[v] = 0.0f; }

  h16* myP = sP + wave * 16 * 64;
  const int nkt = qt + 1;  // key tiles of 64, through the block diagonal

  for (int kt = 0; kt < nkt; ++kt) {
    const int key0 = kt * 64;
    __syncthreads();
    // K tile: contiguous 8 KB  ([key0..+63][0..63] f16) -> 512 uint4, 4/thread
    {
      const uint4* src = (const uint4*)(kh + (size_t)key0 * DK);
      uint4* dst = (uint4*)sK;
#pragma unroll
      for (int j = 0; j < 4; ++j) dst[tid + j * 128] = src[tid + j * 128];
    }
    // V^T tile: 64 feat rows x 64 keys -> 512 uint4, 4/thread
#pragma unroll
    for (int j = 0; j < 4; ++j) {
      int idx = tid + j * 128;
      int f = idx >> 3, c = idx & 7;
      ((uint4*)sV)[idx] = *(const uint4*)(vh + (size_t)f * SEQ + key0 + c * 8);
    }
    if (kt + 1 < nkt)  // CDNA5 global_prefetch_b8 for the next K tile
      __builtin_prefetch(kh + (size_t)(key0 + 64) * DK, 0, 1);
    __syncthreads();

    // scores: 16 rows x 64 keys, four 16-col C fragments
    v8f c4[4] = {v8f{}, v8f{}, v8f{}, v8f{}};
#pragma unroll
    for (int n = 0; n < 4; ++n) {
      v16h b0 = load_b(sK + n * 16 * 64, lane, 64);
      v16h b1 = load_b(sK + n * 16 * 64 + 32, lane, 64);
      c4[n] = WMMA16(qa0, b0, c4[n]);
      c4[n] = WMMA16(qa1, b1, c4[n]);
    }

    // online softmax update (row = v+hi8 within wave's 16-row window)
#pragma unroll
    for (int v = 0; v < 8; ++v) {
      const int rg = qbase + hi8 + v;
      float x0 = (key0 + col      <= rg) ? c4[0][v] : -1e30f;
      float x1 = (key0 + 16 + col <= rg) ? c4[1][v] : -1e30f;
      float x2 = (key0 + 32 + col <= rg) ? c4[2][v] : -1e30f;
      float x3 = (key0 + 48 + col <= rg) ? c4[3][v] : -1e30f;
      float tm = fmaxf(fmaxf(x0, x1), fmaxf(x2, x3));
      tm = fmaxf(tm, __shfl_xor(tm, 1, 32));
      tm = fmaxf(tm, __shfl_xor(tm, 2, 32));
      tm = fmaxf(tm, __shfl_xor(tm, 4, 32));
      tm = fmaxf(tm, __shfl_xor(tm, 8, 32));
      float mn = fmaxf(m[v], tm);
      float corr = __expf(m[v] - mn);
      float p0 = __expf(x0 - mn);
      float p1 = __expf(x1 - mn);
      float p2 = __expf(x2 - mn);
      float p3 = __expf(x3 - mn);
      float rs = (p0 + p1) + (p2 + p3);
      rs += __shfl_xor(rs, 1, 32);
      rs += __shfl_xor(rs, 2, 32);
      rs += __shfl_xor(rs, 4, 32);
      rs += __shfl_xor(rs, 8, 32);
      l[v] = l[v] * corr + rs;
      m[v] = mn;
      o[0][v] *= corr; o[1][v] *= corr; o[2][v] *= corr; o[3][v] *= corr;
      h16* pr = myP + (v + hi8) * 64 + col;
      pr[0]  = (h16)p0;
      pr[16] = (h16)p1;
      pr[32] = (h16)p2;
      pr[48] = (h16)p3;
    }

    // P (16x64 over keys) x V^T tile -> accumulate O
    v16h pa0 = load_a(myP, lane, 64);       // keys 0-31 of tile
    v16h pa1 = load_a(myP + 32, lane, 64);  // keys 32-63
#pragma unroll
    for (int n = 0; n < 4; ++n) {
      v16h bv0 = load_b(sV + n * 16 * 64, lane, 64);
      v16h bv1 = load_b(sV + n * 16 * 64 + 32, lane, 64);
      o[n] = WMMA16(pa0, bv0, o[n]);
      o[n] = WMMA16(pa1, bv1, o[n]);
    }
  }

  // epilogue: normalize and store merged-head fp16 output [B][S][H*dk]
#pragma unroll
  for (int v = 0; v < 8; ++v) {
    const int rowg = qbase + hi8 + v;
    const float inv = 1.0f / l[v];
    size_t base = ((size_t)b * SEQ + rowg) * DMODEL + h * DK + col;
#pragma unroll
    for (int n = 0; n < 4; ++n)
      outp[base + n * 16] = (h16)(o[n][v] * inv);
  }
}

// ---------------------------------------------------------------------------
// Kernel 3: output projection.  out = O @ Wo^T  (fp32 result)
// grid = (32, 16); block = 128.  Same tiling as proj_qkv; A is already fp16.
// ---------------------------------------------------------------------------
__global__ __launch_bounds__(128) void proj_out(
    const h16* __restrict__ A, const float* __restrict__ Wo,
    float* __restrict__ out) {
  __shared__ h16 sA[128 * 64];  // 16 KB
  __shared__ h16 sB[64 * 64];   //  8 KB
  const int mt = blockIdx.x, nt = blockIdx.y;
  const int tid = threadIdx.x;
  const int lane = tid & 31, wave = tid >> 5;

  v8f accA[4] = {v8f{}, v8f{}, v8f{}, v8f{}};
  v8f accB[4] = {v8f{}, v8f{}, v8f{}, v8f{}};

  for (int k0 = 0; k0 < DMODEL; k0 += 64) {
    __syncthreads();
    // A tile: fp16 128x64 (1024 uint4, 8/thread)
#pragma unroll
    for (int j = 0; j < 8; ++j) {
      int idx = tid + j * 128;
      int r = idx >> 3, c = idx & 7;
      ((uint4*)sA)[idx] =
          *(const uint4*)(A + (size_t)(mt * 128 + r) * DMODEL + k0 + c * 8);
    }
    // B tile: Wo fp32 -> fp16
#pragma unroll
    for (int j = 0; j < 8; ++j) {
      int idx = tid + j * 128;
      int r = idx >> 4, c4 = idx & 15;
      float4 f = *(const float4*)(Wo + (size_t)(nt * 64 + r) * DMODEL + k0 + c4 * 4);
      h16* d = sB + r * 64 + c4 * 4;
      d[0] = (h16)f.x; d[1] = (h16)f.y; d[2] = (h16)f.z; d[3] = (h16)f.w;
    }
    __syncthreads();

    const h16* aBase = sA + (size_t)(wave * 32) * 64;
    v16h a00 = load_a(aBase, lane, 64);
    v16h a01 = load_a(aBase + 32, lane, 64);
    v16h a10 = load_a(aBase + 16 * 64, lane, 64);
    v16h a11 = load_a(aBase + 16 * 64 + 32, lane, 64);
#pragma unroll
    for (int n = 0; n < 4; ++n) {
      v16h b0 = load_b(sB + n * 16 * 64, lane, 64);
      v16h b1 = load_b(sB + n * 16 * 64 + 32, lane, 64);
      accA[n] = WMMA16(a00, b0, accA[n]);
      accA[n] = WMMA16(a01, b1, accA[n]);
      accB[n] = WMMA16(a10, b0, accB[n]);
      accB[n] = WMMA16(a11, b1, accB[n]);
    }
  }

  const int hi8 = (lane & 16) ? 8 : 0;
  const int col = lane & 15;
#pragma unroll
  for (int g = 0; g < 2; ++g) {
    const v8f* acc = g ? accB : accA;
#pragma unroll
    for (int v = 0; v < 8; ++v) {
      int rowg = mt * 128 + wave * 32 + g * 16 + hi8 + v;
#pragma unroll
      for (int n = 0; n < 4; ++n)
        out[(size_t)rowg * DMODEL + nt * 64 + n * 16 + col] = acc[n][v];
    }
  }
}

// ---------------------------------------------------------------------------
extern "C" void kernel_launch(void* const* d_in, const int* in_sizes, int n_in,
                              void* d_out, int out_size, void* d_ws, size_t ws_size,
                              hipStream_t stream) {
  const float* x  = (const float*)d_in[0];
  // d_in[1] = mask (causal handled analytically)
  const float* Wq = (const float*)d_in[2];
  const float* Wk = (const float*)d_in[3];
  const float* Wv = (const float*)d_in[4];
  const float* Wo = (const float*)d_in[5];

  char* ws = (char*)d_ws;
  const size_t SZ = (size_t)BATCH * SEQ * DMODEL * sizeof(h16);  // 8 MB
  h16* q   = (h16*)(ws);
  h16* kk  = (h16*)(ws + SZ);
  h16* vt  = (h16*)(ws + 2 * SZ);
  h16* o16 = (h16*)(ws + 3 * SZ);

  dim3 g1(MROWS / 128, DMODEL / 64, 3);
  proj_qkv<<<g1, 128, 0, stream>>>(x, Wq, Wk, Wv, q, kk, vt);

  dim3 g2(BATCH * NH * (SEQ / 64));
  attn<<<g2, 128, 0, stream>>>(q, kk, vt, o16);

  dim3 g3(MROWS / 128, DMODEL / 64);
  proj_out<<<g3, 128, 0, stream>>>(o16, Wo, (float*)d_out);
}
